// JGCNConv_81329500717198
// MI455X (gfx1250) — matrix operand order repaired and checked by reading, not compiled, not run on hardware
//
#include <hip/hip_runtime.h>
#include <hip/hip_bf16.h>

#define DI __device__ __forceinline__

typedef __attribute__((ext_vector_type(2)))  float  v2f;
typedef __attribute__((ext_vector_type(8)))  float  v8f;
typedef __attribute__((ext_vector_type(16))) __bf16 v16bf;

enum { B_ = 2, FIN = 256, FOUT = 256, N = 4096, RDIM = 16 };

union BF16Frag { v16bf v; uint4 q[2]; };

DI unsigned short f2bf(float f) {
  unsigned u = __float_as_uint(f);
  u += 0x7FFFu + ((u >> 16) & 1u);        // round-to-nearest-even
  return (unsigned short)(u >> 16);
}

// ---------------------------------------------------------------------------
// 1) rx[b,o,n] = sum_i R_w[o,i] * x[b,i,n];  sq[b,n] = sum_o rx^2
// ---------------------------------------------------------------------------
__global__ void k_rx_sq(const float* __restrict__ x, const float* __restrict__ Rw,
                        float* __restrict__ rx, float* __restrict__ sq) {
  __shared__ float sR[RDIM * FIN];
  for (int t = threadIdx.x; t < RDIM * FIN; t += blockDim.x) sR[t] = Rw[t];
  __syncthreads();
  int gid = blockIdx.x * blockDim.x + threadIdx.x;   // B*N threads
  int b = gid >> 12, n = gid & (N - 1);
  const float* xb = x + (size_t)b * FIN * N + n;
  float acc[RDIM];
#pragma unroll
  for (int o = 0; o < RDIM; ++o) acc[o] = 0.f;
  for (int i = 0; i < FIN; ++i) {
    float xv = xb[(size_t)i * N];
#pragma unroll
    for (int o = 0; o < RDIM; ++o) acc[o] = fmaf(sR[o * FIN + i], xv, acc[o]);
  }
  float s = 0.f;
#pragma unroll
  for (int o = 0; o < RDIM; ++o) {
    rx[((size_t)b * RDIM + o) * N + n] = acc[o];
    s = fmaf(acc[o], acc[o], s);
  }
  sq[gid] = s;
}

// ---------------------------------------------------------------------------
// 1b) xbf = bf16(x)  (one-time convert; x is reused 256x in the big GEMM)
// ---------------------------------------------------------------------------
__global__ void k_xbf16(const float* __restrict__ x, unsigned short* __restrict__ xbf) {
  size_t g = ((size_t)blockIdx.x * blockDim.x + threadIdx.x) * 4;
  float4 v = *(const float4*)(x + g);
  uint2 u;
  u.x = (unsigned)f2bf(v.x) | ((unsigned)f2bf(v.y) << 16);
  u.y = (unsigned)f2bf(v.z) | ((unsigned)f2bf(v.w) << 16);
  *(uint2*)(xbf + g) = u;
}

// ---------------------------------------------------------------------------
// 2) A0[b,i,j] = exp(2*gram - sq_i - sq_j); gram via f32 WMMA, K=16
// ---------------------------------------------------------------------------
__global__ void k_gram_exp(const float* __restrict__ rx, const float* __restrict__ sq,
                           float* __restrict__ A) {
  int wid  = blockIdx.x * (blockDim.x >> 5) + (threadIdx.x >> 5);
  int lane = threadIdx.x & 31, half = lane >> 4, l = lane & 15;
  int b = wid >> 16, r2 = wid & 65535;
  int i0 = (r2 >> 8) << 4, j0 = (r2 & 255) << 4;
  const float* rxb = rx + (size_t)b * RDIM * N;
  v8f c = {};
#pragma unroll
  for (int k0 = 0; k0 < RDIM; k0 += 4) {
    int kb = k0 + half * 2;                 // lanes16-31 hold K=2,3
    v2f a, bb;
    a[0]  = rxb[(size_t)kb * N + i0 + l];
    a[1]  = rxb[(size_t)(kb + 1) * N + i0 + l];
    bb[0] = rxb[(size_t)kb * N + j0 + l];
    bb[1] = rxb[(size_t)(kb + 1) * N + j0 + l];
    c = __builtin_amdgcn_wmma_f32_16x16x4_f32(false, a, false, bb, (short)0, c,
                                              false, false);
  }
  const float* sqb = sq + b * N;
  float sj = sqb[j0 + l];
#pragma unroll
  for (int r = 0; r < 8; ++r) {
    int m = i0 + r + half * 8;
    float v = __expf(2.f * c[r] - sqb[m] - sj);
    A[((size_t)b * N + m) * N + j0 + l] = v;
  }
}

// ---------------------------------------------------------------------------
// 3/6) deg[row] = sum_j A[row, j]   (one wave per row)
// ---------------------------------------------------------------------------
__global__ void k_rowsum(const float* __restrict__ A, float* __restrict__ deg) {
  int row  = blockIdx.x * (blockDim.x >> 5) + (threadIdx.x >> 5);  // B*N rows
  int lane = threadIdx.x & 31;
  const float* p = A + (size_t)row * N;
  float s = 0.f;
  for (int j = lane; j < N; j += 32) s += p[j];
#pragma unroll
  for (int off = 16; off > 0; off >>= 1) s += __shfl_xor(s, off, 32);
  if (lane == 0) deg[row] = s;
}

// 4/7) dinv = rsqrt(max(deg, 1))
__global__ void k_dinv(const float* __restrict__ deg, float* __restrict__ dinv) {
  int t = blockIdx.x * blockDim.x + threadIdx.x;
  dinv[t] = rsqrtf(fmaxf(deg[t], 1.f));
}

// 5) A = dinv_i * A * dinv_j + old_A   (float4 per thread)
__global__ void k_norm_add(float* __restrict__ A, const float* __restrict__ oldA,
                           const float* __restrict__ dinv) {
  size_t g = ((size_t)blockIdx.x * blockDim.x + threadIdx.x) * 4;
  int b = (int)(g >> 24);
  int i = (int)((g >> 12) & (N - 1));
  int j = (int)(g & (N - 1));
  float di = dinv[(b << 12) + i];
  const float* dj = dinv + (b << 12) + j;
  float4 a = *(float4*)(A + g);
  float4 o = *(const float4*)(oldA + g);
  a.x = fmaf(a.x * di, dj[0], o.x);
  a.y = fmaf(a.y * di, dj[1], o.y);
  a.z = fmaf(a.z * di, dj[2], o.z);
  a.w = fmaf(a.w * di, dj[3], o.w);
  *(float4*)(A + g) = a;
}

// ---------------------------------------------------------------------------
// 8) xa[b,i,m] = sum_n x[b,i,n] * A1[b,m,n]*d_m*d_n  -- bf16 WMMA 16x16x32
//    Final normalization of A is FUSED here (each A row is read by exactly one
//    block), and the scaled A is written back as the second kernel output.
//    X tiles (pre-converted bf16) stream via double-buffered
//    global_load_async_to_lds_b128 + s_wait_asynccnt (ASYNCcnt done-in-order).
// ---------------------------------------------------------------------------
__global__ void __launch_bounds__(256) k_gemm_xa(const unsigned short* __restrict__ xbf,
                                                 float* __restrict__ A,
                                                 const float* __restrict__ dinv,
                                                 float* __restrict__ xa) {
  __shared__ unsigned short Xs[2][FIN * 40];  // 2 x (256 rows x (32+8 pad) bf16)
  __shared__ unsigned short As[16 * 40];      // 16 rows x (32+8 pad) bf16
  __shared__ float Ds[N];                     // dinv for this batch (16 KB)
  int b  = blockIdx.x >> 8;
  int m0 = (blockIdx.x & 255) << 4;
  int t  = threadIdx.x;
  int w = t >> 5, lane = t & 31, half = lane >> 4, l = lane & 15;
  int it0 = w << 1;                           // wave owns i-tiles {2w, 2w+1}
  const unsigned short* xg = xbf + (size_t)b * FIN * N;
  float* Ag = A + (size_t)b * N * N + (size_t)m0 * N;

  // async-copy plan: 4 x b128 per thread per chunk (256 rows x 64B)
  auto issue = [&](int nc, int buf) {
#pragma unroll
    for (int k = 0; k < 4; ++k) {
      int q = k * 256 + t;
      int i = q >> 2, seg = q & 3;
      unsigned lds = (unsigned)(uintptr_t)&Xs[buf][i * 40 + seg * 8];
      const unsigned short* gp = xg + (size_t)i * N + nc + seg * 8;
      asm volatile("global_load_async_to_lds_b128 %0, %1, off"
                   :: "v"(lds), "v"((unsigned long long)(uintptr_t)gp)
                   : "memory");
    }
  };

  issue(0, 0);                                // prefetch first X chunk
  for (int q = t; q < N; q += 256) Ds[q] = dinv[(b << 12) + q];
  __syncthreads();
  float dm = Ds[m0 + (t >> 4)];               // row scale for A staging
  v8f acc[2] = {};

  int buf = 0;
  for (int nc = 0; nc < N; nc += 32, buf ^= 1) {
    if (nc + 32 < N) issue(nc + 32, buf ^ 1); // overlap next X fetch with work
    // stage A chunk: load f32, apply d_m*d_n, write scaled A back, bf16 to LDS
    {
      int m = t >> 4, jj = (t & 15) * 2;
      float2 av = *(float2*)(Ag + (size_t)m * N + nc + jj);
      float s0 = av.x * dm * Ds[nc + jj];
      float s1 = av.y * dm * Ds[nc + jj + 1];
      float2 sv; sv.x = s0; sv.y = s1;
      *(float2*)(Ag + (size_t)m * N + nc + jj) = sv;
      *(unsigned*)(&As[m * 40 + jj]) = (unsigned)f2bf(s0) | ((unsigned)f2bf(s1) << 16);
    }
    if (nc + 32 < N) asm volatile("s_wait_asynccnt 0x4" ::: "memory");
    else             asm volatile("s_wait_asynccnt 0x0" ::: "memory");
    __syncthreads();
    // B operand: B[k][col] = A_scaled[m0+col][nc+k]; elem j -> k = 16*half + j
    BF16Frag bop;
    const unsigned short* br = &As[l * 40 + half * 16];
    bop.q[0] = *(const uint4*)(br);
    bop.q[1] = *(const uint4*)(br + 8);
#pragma unroll
    for (int tt = 0; tt < 2; ++tt) {
      // A operand: row i; elem j -> k = 16*half + j
      int i = ((it0 + tt) << 4) + l;
      const unsigned short* ar = &Xs[buf][i * 40 + half * 16];
      BF16Frag aop;
      aop.q[0] = *(const uint4*)(ar);
      aop.q[1] = *(const uint4*)(ar + 8);
      acc[tt] = __builtin_amdgcn_wmma_f32_16x16x32_bf16(
          false, aop.v, false, bop.v, (short)0, acc[tt], false, false);
    }
    __syncthreads();
  }
#pragma unroll
  for (int tt = 0; tt < 2; ++tt)
#pragma unroll
    for (int r = 0; r < 8; ++r) {
      int i = ((it0 + tt) << 4) + r + half * 8;
      xa[((size_t)b * FIN + i) * N + m0 + l] = acc[tt][r];
    }
}

// ---------------------------------------------------------------------------
// 9) out[b,o,n] = sum_i W_w[o,i] * xa[b,i,n] + W_b[o]   -- f32 WMMA, K=256
// ---------------------------------------------------------------------------
__global__ void k_out(const float* __restrict__ xa, const float* __restrict__ Ww,
                      const float* __restrict__ Wb, float* __restrict__ out) {
  int wid  = blockIdx.x * (blockDim.x >> 5) + (threadIdx.x >> 5);
  int lane = threadIdx.x & 31, half = lane >> 4, l = lane & 15;
  int b = wid >> 12, r2 = wid & 4095;
  int o0 = (r2 >> 8) << 4, n0 = (r2 & 255) << 4;
  const float* xab = xa + (size_t)b * FIN * N;
  v8f c = {};
  for (int k0 = 0; k0 < FIN; k0 += 4) {
    int kb = k0 + half * 2;
    v2f a, bb;
    a[0]  = Ww[(o0 + l) * FIN + kb];
    a[1]  = Ww[(o0 + l) * FIN + kb + 1];
    bb[0] = xab[(size_t)kb * N + n0 + l];
    bb[1] = xab[(size_t)(kb + 1) * N + n0 + l];
    c = __builtin_amdgcn_wmma_f32_16x16x4_f32(false, a, false, bb, (short)0, c,
                                              false, false);
  }
#pragma unroll
  for (int r = 0; r < 8; ++r) {
    int o = o0 + r + half * 8;
    out[((size_t)b * FOUT + o) * N + n0 + l] = c[r] + Wb[o];
  }
}

// ---------------------------------------------------------------------------
extern "C" void kernel_launch(void* const* d_in, const int* in_sizes, int n_in,
                              void* d_out, int out_size, void* d_ws, size_t ws_size,
                              hipStream_t stream) {
  (void)in_sizes; (void)n_in; (void)out_size; (void)ws_size;
  const float* x    = (const float*)d_in[0];   // [B, FIN, N]
  const float* oldA = (const float*)d_in[1];   // [B, N, N]
  const float* Rw   = (const float*)d_in[2];   // [RDIM, FIN]
  const float* Ww   = (const float*)d_in[3];   // [FOUT, FIN]
  const float* Wb   = (const float*)d_in[4];   // [FOUT]

  float* out = (float*)d_out;                  // [B, FOUT, N]
  float* A   = out + (size_t)B_ * FOUT * N;    // [B, N, N] (output #2, built in place)

  float* rx   = (float*)d_ws;                  // [B, RDIM, N]
  float* sq   = rx + (size_t)B_ * RDIM * N;    // [B, N]
  float* deg  = sq + (size_t)B_ * N;           // [B, N]
  float* dinv = deg + (size_t)B_ * N;          // [B, N]
  float* xa   = dinv + (size_t)B_ * N;         // [B, FIN, N]
  unsigned short* xbf = (unsigned short*)(xa + (size_t)B_ * FIN * N); // [B, FIN, N] bf16

  unsigned ew_blocks = (unsigned)((size_t)B_ * N * N / 4 / 256);

  k_rx_sq   <<<(B_ * N) / 256, 256, 0, stream>>>(x, Rw, rx, sq);
  k_xbf16   <<<(B_ * FIN * N) / 4 / 256, 256, 0, stream>>>(x, xbf);
  k_gram_exp<<<(B_ * 256 * 256) / 8, 256, 0, stream>>>(rx, sq, A);
  k_rowsum  <<<(B_ * N) / 8, 256, 0, stream>>>(A, deg);
  k_dinv    <<<(B_ * N) / 256, 256, 0, stream>>>(deg, dinv);
  k_norm_add<<<ew_blocks, 256, 0, stream>>>(A, oldA, dinv);
  k_rowsum  <<<(B_ * N) / 8, 256, 0, stream>>>(A, deg);
  k_dinv    <<<(B_ * N) / 256, 256, 0, stream>>>(deg, dinv);
  k_gemm_xa <<<B_ * (N / 16), 256, 0, stream>>>(xbf, A, dinv, xa);  // scale fused
  k_out     <<<(B_ * (FOUT / 16) * (N / 16)) / 8, 256, 0, stream>>>(xa, Ww, Wb, out);
}
